// RNNModel_7834020348625
// MI455X (gfx1250) — matrix operand order repaired and checked by reading, not compiled, not run on hardware
//
#include <hip/hip_runtime.h>

// ---------------------------------------------------------------------------
// Fused RNN scan for gfx1250 (MI455X, wave32, WMMA, async-to-LDS).
//   h_t = tanh( x_t @ W_ih^T + (b_ih+b_hh) + h_{t-1} @ W_hh^T )
//   out = h_T @ W_fc^T + b_fc
// Path A (ws >= 64MB): pre-convert x -> f16 padded [B][T][64] in d_ws, then
//   scan kernel streams x tiles with GLOBAL_LOAD_ASYNC_TO_LDS_B128 into a
//   4-deep LDS ring (ASYNCcnt), keeping WMMA off the global-latency path.
// Path B (fallback): fused kernel with synchronous x staging.
// Weights live in registers as persistent WMMA B-fragments; h ping-pongs
// through LDS between D-layout (f32 accum) and A-layout (f16 operand).
// ---------------------------------------------------------------------------

typedef _Float16 half8  __attribute__((ext_vector_type(8)));
typedef _Float16 half16 __attribute__((ext_vector_type(16)));
typedef float    f32x8  __attribute__((ext_vector_type(8)));

#define B_SZ 1024
#define T_SZ 512
#define I_SZ 45
#define H_SZ 128
#define O_SZ 45

#define BM      16              // batch rows per block
#define NWAVES  4
#define THREADS (NWAVES * 32)

#define WSTR 136                // LDS row stride (halfs) for K=128 arrays (16B aligned, bank-rotating)
#define XSTR 72                 // LDS row stride (halfs) for K=64  arrays
#define XPAD 64                 // padded K of x in ws (f16)
#define NXBUF 4                 // async x ring depth
#define XBUF_BYTES (BM * XSTR * 2)

__device__ __forceinline__ half16 ld_frag16(const _Float16* p0, const _Float16* p1) {
    half16 r;
    *reinterpret_cast<half8*>(&r)       = *reinterpret_cast<const half8*>(p0); // ds_load_b128
    *(reinterpret_cast<half8*>(&r) + 1) = *reinterpret_cast<const half8*>(p1); // ds_load_b128
    return r;
}

// A-matrix 16x32 f16 fragment (ISA 7.12.2): lane<16 -> row=lane, K=k0+0..7 / k0+16..23
//                                           lane>=16 -> row=lane-16, K=k0+8..15 / k0+24..31
__device__ __forceinline__ half16 a_frag(const _Float16* base, int stride, int k0, int lane) {
    const int row = lane & 15;
    const int kb  = k0 + ((lane >> 4) << 3);
    const _Float16* p = base + row * stride + kb;
    return ld_frag16(p, p + 16);
}

// B-matrix 32x16 f16 fragment: lane<16 -> N=n0+lane, K=k0+0..7 / k0+16..23 ; lane>=16 -> K=k0+8..15 / k0+24..31
__device__ __forceinline__ half16 b_frag(const _Float16* base, int stride, int n0, int k0, int lane) {
    const int n  = n0 + (lane & 15);
    const int kb = k0 + ((lane >> 4) << 3);
    const _Float16* p = base + n * stride + kb;
    return ld_frag16(p, p + 16);
}

__device__ __forceinline__ f32x8 wmma_f16(half16 a, half16 b, f32x8 c) {
    return __builtin_amdgcn_wmma_f32_16x16x32_f16(false, a, false, b, (short)0, c, false, false);
}

__device__ __forceinline__ float tanh_fast(float x) {
#if __has_builtin(__builtin_amdgcn_tanhf)
    return __builtin_amdgcn_tanhf(x);          // v_tanh_f32 (gfx1250 TRANS op)
#elif __has_builtin(__builtin_amdgcn_tanh_f32)
    return __builtin_amdgcn_tanh_f32(x);
#else
    x = fminf(fmaxf(x, -15.0f), 15.0f);
    const float e = __expf(2.0f * x);
    return (e - 1.0f) * __builtin_amdgcn_rcpf(e + 1.0f);
#endif
}

// async copy of 16B global -> LDS, tracked by ASYNCcnt (CDNA5 VGLOBAL GVS form)
__device__ __forceinline__ void async_load_b128(unsigned lds_addr, unsigned gvofs, const void* sbase) {
    asm volatile("global_load_async_to_lds_b128 %0, %1, %2"
                 :: "v"(lds_addr), "v"(gvofs), "s"(sbase) : "memory");
}

// ---------------------------------------------------------------------------
// Kernel 0: x f32 [B][T][45] -> f16 [B][T][64] (zero-padded K) in workspace
// ---------------------------------------------------------------------------
__global__ __launch_bounds__(256)
void convert_x_kernel(const float* __restrict__ x, _Float16* __restrict__ x16) {
    const long tid = (long)blockIdx.x * 256 + threadIdx.x;   // over B*T*8
    if (tid >= (long)B_SZ * T_SZ * 8) return;
    const long bt = tid >> 3;
    const int chunk = (int)(tid & 7);
    const float* xr = x + bt * I_SZ;
    half8 v;
#pragma unroll
    for (int j = 0; j < 8; ++j) {
        const int k = chunk * 8 + j;
        v[j] = (k < I_SZ) ? (_Float16)xr[k] : (_Float16)0.0f;
    }
    *reinterpret_cast<half8*>(x16 + bt * XPAD + chunk * 8) = v;
}

// ---------------------------------------------------------------------------
// Kernel 1 (path A): scan with async-to-LDS x streaming
// ---------------------------------------------------------------------------
__global__ __launch_bounds__(THREADS)
void rnn_scan_async(const _Float16* __restrict__ x16,
                    const float* __restrict__ W_ih, const float* __restrict__ b_ih,
                    const float* __restrict__ W_hh, const float* __restrict__ b_hh,
                    const float* __restrict__ W_fc, const float* __restrict__ b_fc,
                    float* __restrict__ out)
{
    __shared__ _Float16 s_Whh[H_SZ * WSTR];
    __shared__ _Float16 s_Wih[H_SZ * XSTR];
    __shared__ _Float16 s_Wfc[48 * WSTR];
    __shared__ _Float16 s_h[2][BM * WSTR];
    __shared__ _Float16 s_x[NXBUF][BM * XSTR];   // async x ring
    __shared__ float    s_bias[H_SZ];
    __shared__ float    s_bfc[48];

    const int tid  = threadIdx.x;
    const int lane = tid & 31;
    const int wave = tid >> 5;
    const int b0   = blockIdx.x * BM;

    // ---- stage weights into LDS ----
    for (int idx = tid; idx < H_SZ * H_SZ; idx += THREADS) {
        const int n = idx >> 7, k = idx & 127;
        s_Whh[n * WSTR + k] = (_Float16)W_hh[idx];
    }
    for (int idx = tid; idx < H_SZ * 64; idx += THREADS) {
        const int n = idx >> 6, k = idx & 63;
        s_Wih[n * XSTR + k] = (k < I_SZ) ? (_Float16)W_ih[n * I_SZ + k] : (_Float16)0.0f;
    }
    for (int idx = tid; idx < 48 * H_SZ; idx += THREADS) {
        const int n = idx >> 7, k = idx & 127;
        s_Wfc[n * WSTR + k] = (n < O_SZ) ? (_Float16)W_fc[n * H_SZ + k] : (_Float16)0.0f;
    }
    for (int idx = tid; idx < H_SZ; idx += THREADS) s_bias[idx] = b_ih[idx] + b_hh[idx];
    if (tid < 48) s_bfc[tid] = (tid < O_SZ) ? b_fc[tid] : 0.0f;
    for (int idx = tid; idx < BM * WSTR; idx += THREADS) s_h[0][idx] = (_Float16)0.0f;

    // ---- async x prologue: each thread owns one 16B chunk of the 16x64 tile ----
    const int xr    = tid >> 3;          // batch row within tile (0..15)
    const int chunk = tid & 7;           // 8 halfs (16B) chunk within row
    const unsigned ldsx  = (unsigned)(unsigned long long)(&s_x[0][0]) +
                           (unsigned)((xr * XSTR + chunk * 8) * 2);
    const unsigned gbase = (unsigned)((b0 + xr) * T_SZ) * (XPAD * 2u) + chunk * 16u;
#pragma unroll
    for (int s = 0; s < NXBUF; ++s)
        async_load_b128(ldsx + s * XBUF_BYTES, gbase + (unsigned)s * (XPAD * 2u), x16);

    __syncthreads();

    // ---- per-wave persistent B fragments (wave owns columns n0a..n0a+31) ----
    const int n0a = wave * 32, n0b = n0a + 16;
    half16 Bih_a[2], Bih_b[2], Bhh_a[4], Bhh_b[4];
#pragma unroll
    for (int kf = 0; kf < 2; ++kf) {
        Bih_a[kf] = b_frag(s_Wih, XSTR, n0a, kf * 32, lane);
        Bih_b[kf] = b_frag(s_Wih, XSTR, n0b, kf * 32, lane);
    }
#pragma unroll
    for (int kf = 0; kf < 4; ++kf) {
        Bhh_a[kf] = b_frag(s_Whh, WSTR, n0a, kf * 32, lane);
        Bhh_b[kf] = b_frag(s_Whh, WSTR, n0b, kf * 32, lane);
    }
    const int ccol = lane & 15;
    const int crow = (lane >> 4) << 3;
    const float bias_a = s_bias[n0a + ccol];
    const float bias_b = s_bias[n0b + ccol];

    // ---- sequential scan over T ----
    for (int t = 0; t < T_SZ; ++t) {
        asm volatile("s_wait_asynccnt 0x3" ::: "memory");  // x tile for step t landed
        __syncthreads();                                   // ..in every wave; h[p] visible

        const int p = t & 1;
        const _Float16* hb = s_h[p];
        const _Float16* xb = s_x[t & (NXBUF - 1)];
        f32x8 acc_a = {}, acc_b = {};
#pragma unroll
        for (int kf = 0; kf < 2; ++kf) {       // input contribution: x_t @ W_ih^T
            const half16 a = a_frag(xb, XSTR, kf * 32, lane);
            acc_a = wmma_f16(a, Bih_a[kf], acc_a);
            acc_b = wmma_f16(a, Bih_b[kf], acc_b);
        }
#pragma unroll
        for (int kf = 0; kf < 4; ++kf) {       // recurrent: h @ W_hh^T
            const half16 a = a_frag(hb, WSTR, kf * 32, lane);
            acc_a = wmma_f16(a, Bhh_a[kf], acc_a);
            acc_b = wmma_f16(a, Bhh_b[kf], acc_b);
        }
        _Float16* hn = s_h[p ^ 1];
#pragma unroll
        for (int r = 0; r < 8; ++r) {
            hn[(crow + r) * WSTR + n0a + ccol] = (_Float16)tanh_fast(acc_a[r] + bias_a);
            hn[(crow + r) * WSTR + n0b + ccol] = (_Float16)tanh_fast(acc_b[r] + bias_b);
        }
        __syncthreads();                       // all waves done reading xb and h[p]
        // refill ring slot (t%4) with step t+4 (clamped re-read keeps counts uniform)
        const int ts = (t + NXBUF < T_SZ) ? (t + NXBUF) : t;
        async_load_b128(ldsx + (t & (NXBUF - 1)) * XBUF_BYTES,
                        gbase + (unsigned)ts * (XPAD * 2u), x16);
    }
    asm volatile("s_wait_asynccnt 0x0" ::: "memory");
    __syncthreads();

    // ---- final FC: out = h_T @ W_fc^T + b_fc  (T even -> h_T in buffer 0) ----
    const _Float16* hl = s_h[0];
    if (wave < 3) {
        const int n0 = wave * 16;
        f32x8 acc = {};
#pragma unroll
        for (int kf = 0; kf < 4; ++kf) {
            const half16 a = a_frag(hl, WSTR, kf * 32, lane);
            const half16 b = b_frag(s_Wfc, WSTR, n0, kf * 32, lane);
            acc = wmma_f16(a, b, acc);
        }
        const int col = n0 + ccol;
        if (col < O_SZ) {
            const float bf = s_bfc[col];
#pragma unroll
            for (int r = 0; r < 8; ++r)
                out[(long)(b0 + crow + r) * O_SZ + col] = acc[r] + bf;
        }
    }
}

// ---------------------------------------------------------------------------
// Kernel 2 (path B fallback): fused scan with synchronous x staging
// ---------------------------------------------------------------------------
__global__ __launch_bounds__(THREADS)
void rnn_scan_sync(const float* __restrict__ x,    const float* __restrict__ W_ih,
                   const float* __restrict__ b_ih, const float* __restrict__ W_hh,
                   const float* __restrict__ b_hh, const float* __restrict__ W_fc,
                   const float* __restrict__ b_fc, float* __restrict__ out)
{
    __shared__ _Float16 s_Whh[H_SZ * WSTR];
    __shared__ _Float16 s_Wih[H_SZ * XSTR];
    __shared__ _Float16 s_Wfc[48 * WSTR];
    __shared__ _Float16 s_h[2][BM * WSTR];
    __shared__ _Float16 s_x[BM * XSTR];
    __shared__ float    s_bias[H_SZ];
    __shared__ float    s_bfc[48];

    const int tid  = threadIdx.x;
    const int lane = tid & 31;
    const int wave = tid >> 5;
    const int b0   = blockIdx.x * BM;

    for (int idx = tid; idx < H_SZ * H_SZ; idx += THREADS) {
        const int n = idx >> 7, k = idx & 127;
        s_Whh[n * WSTR + k] = (_Float16)W_hh[idx];
    }
    for (int idx = tid; idx < H_SZ * 64; idx += THREADS) {
        const int n = idx >> 6, k = idx & 63;
        s_Wih[n * XSTR + k] = (k < I_SZ) ? (_Float16)W_ih[n * I_SZ + k] : (_Float16)0.0f;
    }
    for (int idx = tid; idx < 48 * H_SZ; idx += THREADS) {
        const int n = idx >> 7, k = idx & 127;
        s_Wfc[n * WSTR + k] = (n < O_SZ) ? (_Float16)W_fc[n * H_SZ + k] : (_Float16)0.0f;
    }
    for (int idx = tid; idx < H_SZ; idx += THREADS) s_bias[idx] = b_ih[idx] + b_hh[idx];
    if (tid < 48) s_bfc[tid] = (tid < O_SZ) ? b_fc[tid] : 0.0f;
    for (int idx = tid; idx < BM * WSTR; idx += THREADS) s_h[0][idx] = (_Float16)0.0f;
    __syncthreads();

    const int n0a = wave * 32, n0b = n0a + 16;
    half16 Bih_a[2], Bih_b[2], Bhh_a[4], Bhh_b[4];
#pragma unroll
    for (int kf = 0; kf < 2; ++kf) {
        Bih_a[kf] = b_frag(s_Wih, XSTR, n0a, kf * 32, lane);
        Bih_b[kf] = b_frag(s_Wih, XSTR, n0b, kf * 32, lane);
    }
#pragma unroll
    for (int kf = 0; kf < 4; ++kf) {
        Bhh_a[kf] = b_frag(s_Whh, WSTR, n0a, kf * 32, lane);
        Bhh_b[kf] = b_frag(s_Whh, WSTR, n0b, kf * 32, lane);
    }
    const int ccol = lane & 15;
    const int crow = (lane >> 4) << 3;
    const float bias_a = s_bias[n0a + ccol];
    const float bias_b = s_bias[n0b + ccol];

    for (int t = 0; t < T_SZ; ++t) {
        float xv[(BM * 64) / THREADS];
#pragma unroll
        for (int j = 0; j < (BM * 64) / THREADS; ++j) {
            const int idx = tid + j * THREADS;
            const int r = idx >> 6, k = idx & 63;
            xv[j] = (k < I_SZ) ? x[(((long)(b0 + r) * T_SZ) + t) * (long)I_SZ + k] : 0.0f;
        }
        __syncthreads();
#pragma unroll
        for (int j = 0; j < (BM * 64) / THREADS; ++j) {
            const int idx = tid + j * THREADS;
            const int r = idx >> 6, k = idx & 63;
            s_x[r * XSTR + k] = (_Float16)xv[j];
        }
        if (wave == 0 && lane < BM && (t + 1) < T_SZ) {
            const float* np = &x[(((long)(b0 + lane) * T_SZ) + (t + 1)) * (long)I_SZ];
            __builtin_prefetch(np, 0, 0);
            __builtin_prefetch(np + 32, 0, 0);
        }
        __syncthreads();

        const int p = t & 1;
        const _Float16* hb = s_h[p];
        f32x8 acc_a = {}, acc_b = {};
#pragma unroll
        for (int kf = 0; kf < 2; ++kf) {
            const half16 a = a_frag(s_x, XSTR, kf * 32, lane);
            acc_a = wmma_f16(a, Bih_a[kf], acc_a);
            acc_b = wmma_f16(a, Bih_b[kf], acc_b);
        }
#pragma unroll
        for (int kf = 0; kf < 4; ++kf) {
            const half16 a = a_frag(hb, WSTR, kf * 32, lane);
            acc_a = wmma_f16(a, Bhh_a[kf], acc_a);
            acc_b = wmma_f16(a, Bhh_b[kf], acc_b);
        }
        _Float16* hn = s_h[p ^ 1];
#pragma unroll
        for (int r = 0; r < 8; ++r) {
            hn[(crow + r) * WSTR + n0a + ccol] = (_Float16)tanh_fast(acc_a[r] + bias_a);
            hn[(crow + r) * WSTR + n0b + ccol] = (_Float16)tanh_fast(acc_b[r] + bias_b);
        }
    }
    __syncthreads();

    const _Float16* hl = s_h[0];
    if (wave < 3) {
        const int n0 = wave * 16;
        f32x8 acc = {};
#pragma unroll
        for (int kf = 0; kf < 4; ++kf) {
            const half16 a = a_frag(hl, WSTR, kf * 32, lane);
            const half16 b = b_frag(s_Wfc, WSTR, n0, kf * 32, lane);
            acc = wmma_f16(a, b, acc);
        }
        const int col = n0 + ccol;
        if (col < O_SZ) {
            const float bf = s_bfc[col];
#pragma unroll
            for (int r = 0; r < 8; ++r)
                out[(long)(b0 + crow + r) * O_SZ + col] = acc[r] + bf;
        }
    }
}

extern "C" void kernel_launch(void* const* d_in, const int* in_sizes, int n_in,
                              void* d_out, int out_size, void* d_ws, size_t ws_size,
                              hipStream_t stream) {
    const float* x    = (const float*)d_in[0];
    const float* W_ih = (const float*)d_in[1];
    const float* b_ih = (const float*)d_in[2];
    const float* W_hh = (const float*)d_in[3];
    const float* b_hh = (const float*)d_in[4];
    const float* W_fc = (const float*)d_in[5];
    const float* b_fc = (const float*)d_in[6];
    float* out = (float*)d_out;

    const size_t need = (size_t)B_SZ * T_SZ * XPAD * sizeof(_Float16);  // 64 MB
    if (ws_size >= need) {
        _Float16* x16 = (_Float16*)d_ws;
        hipLaunchKernelGGL(convert_x_kernel, dim3((B_SZ * T_SZ * 8) / 256), dim3(256), 0, stream,
                           x, x16);
        hipLaunchKernelGGL(rnn_scan_async, dim3(B_SZ / BM), dim3(THREADS), 0, stream,
                           x16, W_ih, b_ih, W_hh, b_hh, W_fc, b_fc, out);
    } else {
        hipLaunchKernelGGL(rnn_scan_sync, dim3(B_SZ / BM), dim3(THREADS), 0, stream,
                           x, W_ih, b_ih, W_hh, b_hh, W_fc, b_fc, out);
    }
    (void)in_sizes; (void)n_in; (void)out_size;
}
